// MultiHeadAttention_21646635171970
// MI455X (gfx1250) — compile-verified, matrix operand order checked by reading, not run on hardware
//
#include <hip/hip_runtime.h>

// ---------------------------------------------------------------------------
// Multi-head attention forward for MI455X (gfx1250, wave32, WMMA).
//   D=1024, H=16, HD=64, B=8, S=1024.
//   d_out = [ out (B*S*D fp32) | probs (B*H*S*S fp32) ]
// Strategy:
//   * all GEMMs in f16 WMMA (v_wmma_f32_16x16x32_f16), fp32 accumulate
//   * V stored transposed [B,H,HD,S] so P@V B-fragments are K-contiguous
//     (no LDS gather path; everything is global b128 loads)
//   * two-pass softmax; 512MB probs written once with nontemporal stores so
//     the 192MB L2 keeps Q/K/V resident across the 64 query tiles per head
// ---------------------------------------------------------------------------

#define DD 1024
#define HH 16
#define HD 64
#define BB 8
#define SS 1024

typedef __attribute__((ext_vector_type(16))) _Float16 v16h;
typedef __attribute__((ext_vector_type(8)))  _Float16 v8h;
typedef __attribute__((ext_vector_type(8)))  float    v8f;

union Frag {
    v16h v;
    v8h  h[2];
    _Float16 e[16];
};

// Load a 16x32 f16 A/B fragment whose K-dim is contiguous in memory.
// Per ISA 7.12.2: lanes 0-15 hold K=[0..7],[16..23], lanes 16-31 hold
// K=[8..15],[24..31]  (g = 0 or 8 halfs = 0 or 16 bytes).
__device__ __forceinline__ Frag load_frag_kcontig(const _Float16* rowk, int g) {
    Frag f;
    f.h[0] = *(const v8h*)(rowk + g);
    f.h[1] = *(const v8h*)(rowk + 16 + g);
    return f;
}

__device__ __forceinline__ v8f wmma_f16(const Frag& a, const Frag& b, v8f c) {
    return __builtin_amdgcn_wmma_f32_16x16x32_f16(false, a.v, false, b.v,
                                                  (short)0, c, false, false);
}

// ---------------------------------------------------------------------------
// Kernel 0: fp32 -> f16 cast, 8 elements/thread (two b128 loads, one b128 st).
// ---------------------------------------------------------------------------
__global__ void cast_f32_to_f16_kernel(const float* __restrict__ src,
                                       _Float16* __restrict__ dst, int n) {
    int i = (blockIdx.x * blockDim.x + threadIdx.x) * 8;
    if (i < n) {
        float4 a = *(const float4*)(src + i);
        float4 b = *(const float4*)(src + i + 4);
        v8h d;
        d[0] = (_Float16)a.x; d[1] = (_Float16)a.y;
        d[2] = (_Float16)a.z; d[3] = (_Float16)a.w;
        d[4] = (_Float16)b.x; d[5] = (_Float16)b.y;
        d[6] = (_Float16)b.z; d[7] = (_Float16)b.w;
        *(v8h*)(dst + i) = d;
    }
}

// ---------------------------------------------------------------------------
// Kernel 1: fused QKV projection.  Y = x @ W^T + b.
//   Q, K stored f16 as [B,H,S,HD]; V stored TRANSPOSED as [B,H,HD,S].
// grid = (B*S/16, 16, 3), block = 128 (4 waves; each wave owns one N-tile).
// ---------------------------------------------------------------------------
__global__ void qkv_proj_kernel(const _Float16* __restrict__ xh,
                                const _Float16* __restrict__ wq,
                                const _Float16* __restrict__ wk,
                                const _Float16* __restrict__ wv,
                                const float* __restrict__ bq,
                                const float* __restrict__ bk,
                                const float* __restrict__ bv,
                                _Float16* __restrict__ qh,
                                _Float16* __restrict__ kh,
                                _Float16* __restrict__ vth) {
    const int lane  = threadIdx.x & 31;
    const int wave  = threadIdx.x >> 5;
    const int mtile = blockIdx.x;               // 0..511  (B*S/16)
    const int ntile = blockIdx.y * 4 + wave;    // 0..63   (D/16)

    const _Float16* W;
    const float*    bias;
    _Float16*       dst;
    const bool transpose_v = (blockIdx.z == 2);
    if (blockIdx.z == 0)      { W = wq; bias = bq; dst = qh;  }
    else if (blockIdx.z == 1) { W = wk; bias = bk; dst = kh;  }
    else                      { W = wv; bias = bv; dst = vth; }

    const int g       = (lane < 16) ? 0 : 8;
    const int halfoff = (lane < 16) ? 0 : 8;
    const int arow    = mtile * 16 + (lane & 15);
    const int bcol    = ntile * 16 + (lane & 15);   // output column
    const _Float16* ap = xh + (size_t)arow * DD;
    const _Float16* bp = W  + (size_t)bcol * DD;    // W[n,k] contiguous in k

    v8f c = {};
    #pragma unroll 4
    for (int k = 0; k < DD; k += 32) {
        Frag a = load_frag_kcontig(ap + k, g);
        Frag b = load_frag_kcontig(bp + k, g);
        c = wmma_f16(a, b, c);
    }

    const float bn = bias[bcol];
    const int hh = bcol >> 6;          // head
    const int hd = bcol & (HD - 1);    // dim within head
    #pragma unroll
    for (int i = 0; i < 8; ++i) {
        int m  = mtile * 16 + i + halfoff;       // flattened b*S+s
        int bi = m >> 10;
        int si = m & (SS - 1);
        _Float16 val = (_Float16)(c[i] + bn);
        if (transpose_v)
            dst[((size_t)(bi * HH + hh) * HD + hd) * SS + si] = val;  // [B,H,HD,S]
        else
            dst[((size_t)(bi * HH + hh) * SS + si) * HD + hd] = val;  // [B,H,S,HD]
    }
}

// ---------------------------------------------------------------------------
// Kernel 2: softmax row stats (online max / sum-exp), per 16-query tile.
// grid = B*H*(S/16)/4 = 2048 blocks of 128 (4 independent waves).
// ---------------------------------------------------------------------------
__global__ void attn_stats_kernel(const _Float16* __restrict__ qh,
                                  const _Float16* __restrict__ kh,
                                  float* __restrict__ rowmax,
                                  float* __restrict__ rowsum) {
    const int lane = threadIdx.x & 31;
    const int wave = threadIdx.x >> 5;
    const int idx  = blockIdx.x * 4 + wave;     // 0..8191
    const int qt   = idx & 63;                  // query tile within sequence
    const int bh   = idx >> 6;                  // (b*H + h)

    const int g = (lane < 16) ? 0 : 8;
    const _Float16* qp = qh + ((size_t)bh * SS + qt * 16 + (lane & 15)) * HD;
    Frag aq0 = load_frag_kcontig(qp,      g);   // d = 0..31
    Frag aq1 = load_frag_kcontig(qp + 32, g);   // d = 32..63

    float rm[8], rs[8];
    #pragma unroll
    for (int i = 0; i < 8; ++i) { rm[i] = -3.0e38f; rs[i] = 0.0f; }

    const _Float16* kbase = kh + (size_t)bh * SS * HD;
    for (int j = 0; j < SS / 16; ++j) {
        const _Float16* kp = kbase + ((size_t)j * 16 + (lane & 15)) * HD;
        // prefetch next key tile while this one computes
        __builtin_prefetch(kbase + ((size_t)((j + 1) & 63) * 16 + (lane & 15)) * HD, 0, 1);
        Frag bk0 = load_frag_kcontig(kp,      g);
        Frag bk1 = load_frag_kcontig(kp + 32, g);
        v8f c = {};
        c = wmma_f16(aq0, bk0, c);
        c = wmma_f16(aq1, bk1, c);

        #pragma unroll
        for (int i = 0; i < 8; ++i) {
            float sc = c[i] * 0.125f;           // 1/sqrt(HD)
            float tm = sc;
            #pragma unroll
            for (int off = 1; off < 16; off <<= 1)
                tm = fmaxf(tm, __shfl_xor(tm, off, 32));
            float nm = fmaxf(rm[i], tm);
            float p  = __expf(sc - nm);
            #pragma unroll
            for (int off = 1; off < 16; off <<= 1)
                p += __shfl_xor(p, off, 32);
            rs[i] = rs[i] * __expf(rm[i] - nm) + p;
            rm[i] = nm;
        }
    }

    if ((lane & 15) == 0) {
        size_t base = (size_t)bh * SS + qt * 16 + ((lane == 16) ? 8 : 0);
        #pragma unroll
        for (int i = 0; i < 8; ++i) {
            rowmax[base + i] = rm[i];
            rowsum[base + i] = rs[i];
        }
    }
}

// ---------------------------------------------------------------------------
// Kernel 3: recompute scores, emit normalized probs (nontemporal fp32 to
// d_out), and accumulate ctx = P @ V with WMMA.  Probs are transposed
// D-layout -> A-layout through a small LDS tile; V fragments come straight
// from the transposed V in global memory (K-contiguous b128 loads).
// ---------------------------------------------------------------------------
__global__ void attn_ctx_kernel(const _Float16* __restrict__ qh,
                                const _Float16* __restrict__ kh,
                                const _Float16* __restrict__ vth,
                                const float* __restrict__ rowmax,
                                const float* __restrict__ rowsum,
                                float* __restrict__ probs_out,
                                _Float16* __restrict__ ctxh) {
    __shared__ alignas(16) _Float16 ph[4][16][32];   // probs tile per wave

    const int lane = threadIdx.x & 31;
    const int wave = threadIdx.x >> 5;
    const int idx  = blockIdx.x * 4 + wave;
    const int qt   = idx & 63;
    const int bh   = idx >> 6;
    const int h    = bh & (HH - 1);
    const int b    = bh >> 4;

    const int g       = (lane < 16) ? 0 : 8;
    const int halfoff = (lane < 16) ? 0 : 8;

    const _Float16* qp = qh + ((size_t)bh * SS + qt * 16 + (lane & 15)) * HD;
    Frag aq0 = load_frag_kcontig(qp,      g);
    Frag aq1 = load_frag_kcontig(qp + 32, g);

    float rm[8], ri[8];
    {
        size_t sbase = (size_t)bh * SS + qt * 16 + halfoff;
        #pragma unroll
        for (int i = 0; i < 8; ++i) {
            rm[i] = rowmax[sbase + i];
            ri[i] = 1.0f / rowsum[sbase + i];
        }
    }

    v8f zero = {};
    v8f acc[4] = {zero, zero, zero, zero};   // ctx 16 x 64 = 4 N-tiles

    const _Float16* kbase = kh  + (size_t)bh * SS * HD;
    const _Float16* vbase = vth + (size_t)bh * HD * SS;   // [HD][S]

    for (int jp = 0; jp < SS / 32; ++jp) {
        // Two 16-key score subtiles -> probs (nontemporal fp32 out + f16 LDS).
        #pragma unroll
        for (int sub = 0; sub < 2; ++sub) {
            const int j = jp * 2 + sub;
            const _Float16* kp = kbase + ((size_t)j * 16 + (lane & 15)) * HD;
            __builtin_prefetch(kbase + ((size_t)((j + 1) & 63) * 16 + (lane & 15)) * HD, 0, 1);
            Frag bk0 = load_frag_kcontig(kp,      g);
            Frag bk1 = load_frag_kcontig(kp + 32, g);
            v8f c = {};
            c = wmma_f16(aq0, bk0, c);
            c = wmma_f16(aq1, bk1, c);

            #pragma unroll
            for (int i = 0; i < 8; ++i) {
                float p = __expf(c[i] * 0.125f - rm[i]) * ri[i];
                int qrow = qt * 16 + i + halfoff;
                // streaming write: never re-read -> keep it out of L2
                __builtin_nontemporal_store(
                    p, &probs_out[((size_t)bh * SS + qrow) * SS + j * 16 + (lane & 15)]);
                ph[wave][i + halfoff][sub * 16 + (lane & 15)] = (_Float16)p;
            }
        }

        // Probs as A fragment (transpose via LDS), K = 32 keys.
        Frag pa;
        pa.h[0] = *(const v8h*)&ph[wave][lane & 15][g];
        pa.h[1] = *(const v8h*)&ph[wave][lane & 15][16 + g];

        // ctx accumulation: B[k,n] = V^T[d = n][key jp*32 + k], K-contiguous.
        #pragma unroll
        for (int nt = 0; nt < 4; ++nt) {
            const int d = nt * 16 + (lane & 15);
            Frag bv = load_frag_kcontig(vbase + (size_t)d * SS + jp * 32, g);
            acc[nt] = wmma_f16(pa, bv, acc[nt]);
        }
    }

    // Store ctx (recombined heads) as f16 [B,S,D].
    #pragma unroll
    for (int nt = 0; nt < 4; ++nt) {
        const int col = h * HD + nt * 16 + (lane & 15);
        #pragma unroll
        for (int i = 0; i < 8; ++i) {
            int srow = qt * 16 + i + halfoff;
            ctxh[((size_t)b * SS + srow) * DD + col] = (_Float16)acc[nt][i];
        }
    }
}

// ---------------------------------------------------------------------------
// Kernel 4: output projection  out = ctx @ Wf^T + bf  (fp32, nontemporal).
// grid = (512, 16), block = 128.
// ---------------------------------------------------------------------------
__global__ void out_proj_kernel(const _Float16* __restrict__ ctxh,
                                const _Float16* __restrict__ wf,
                                const float* __restrict__ bf,
                                float* __restrict__ out) {
    const int lane  = threadIdx.x & 31;
    const int wave  = threadIdx.x >> 5;
    const int mtile = blockIdx.x;
    const int ntile = blockIdx.y * 4 + wave;

    const int g       = (lane < 16) ? 0 : 8;
    const int halfoff = (lane < 16) ? 0 : 8;
    const int arow    = mtile * 16 + (lane & 15);
    const int bcol    = ntile * 16 + (lane & 15);
    const _Float16* ap = ctxh + (size_t)arow * DD;
    const _Float16* bp = wf   + (size_t)bcol * DD;

    v8f c = {};
    #pragma unroll 4
    for (int k = 0; k < DD; k += 32) {
        Frag a = load_frag_kcontig(ap + k, g);
        Frag b = load_frag_kcontig(bp + k, g);
        c = wmma_f16(a, b, c);
    }

    const float bn = bf[bcol];
    #pragma unroll
    for (int i = 0; i < 8; ++i) {
        int m = mtile * 16 + i + halfoff;
        __builtin_nontemporal_store(c[i] + bn, &out[(size_t)m * DD + bcol]);
    }
}

// ---------------------------------------------------------------------------
// Host launch
// ---------------------------------------------------------------------------
extern "C" void kernel_launch(void* const* d_in, const int* in_sizes, int n_in,
                              void* d_out, int out_size, void* d_ws, size_t ws_size,
                              hipStream_t stream) {
    const float* x  = (const float*)d_in[0];
    const float* Wq = (const float*)d_in[1];
    const float* bq = (const float*)d_in[2];
    const float* Wk = (const float*)d_in[3];
    const float* bk = (const float*)d_in[4];
    const float* Wv = (const float*)d_in[5];
    const float* bv = (const float*)d_in[6];
    const float* Wf = (const float*)d_in[7];
    const float* bf = (const float*)d_in[8];

    const size_t NX = (size_t)BB * SS * DD;   // 8,388,608
    const size_t NW = (size_t)DD * DD;        // 1,048,576

    // Workspace carve-up (~90 MB, all f16 except stats).
    _Float16* p    = (_Float16*)d_ws;
    _Float16* xh   = p; p += NX;
    _Float16* wqh  = p; p += NW;
    _Float16* wkh  = p; p += NW;
    _Float16* wvh  = p; p += NW;
    _Float16* wfh  = p; p += NW;
    _Float16* qh   = p; p += NX;
    _Float16* kh   = p; p += NX;
    _Float16* vth  = p; p += NX;   // V transposed: [B,H,HD,S]
    _Float16* ctxh = p; p += NX;
    float* rowmax = (float*)p;
    float* rowsum = rowmax + (size_t)BB * HH * SS;

    float* out_p   = (float*)d_out;
    float* probs_p = out_p + NX;

    // 0) casts (8 elems/thread)
    cast_f32_to_f16_kernel<<<(int)(NX / 8 / 256), 256, 0, stream>>>(x,  xh,  (int)NX);
    cast_f32_to_f16_kernel<<<(int)(NW / 8 / 256), 256, 0, stream>>>(Wq, wqh, (int)NW);
    cast_f32_to_f16_kernel<<<(int)(NW / 8 / 256), 256, 0, stream>>>(Wk, wkh, (int)NW);
    cast_f32_to_f16_kernel<<<(int)(NW / 8 / 256), 256, 0, stream>>>(Wv, wvh, (int)NW);
    cast_f32_to_f16_kernel<<<(int)(NW / 8 / 256), 256, 0, stream>>>(Wf, wfh, (int)NW);

    // 1) QKV projection (V transposed on store)
    qkv_proj_kernel<<<dim3(BB * SS / 16, DD / 64, 3), 128, 0, stream>>>(
        xh, wqh, wkh, wvh, bq, bk, bv, qh, kh, vth);

    // 2) softmax stats
    attn_stats_kernel<<<BB * HH * (SS / 16) / 4, 128, 0, stream>>>(
        qh, kh, rowmax, rowsum);

    // 3) probs + ctx
    attn_ctx_kernel<<<BB * HH * (SS / 16) / 4, 128, 0, stream>>>(
        qh, kh, vth, rowmax, rowsum, probs_p, ctxh);

    // 4) output projection
    out_proj_kernel<<<dim3(BB * SS / 16, DD / 64), 128, 0, stream>>>(
        ctxh, wfh, bf, out_p);
}
